// BertBlock_34797825032382
// MI455X (gfx1250) — matrix operand order repaired and checked by reading, not compiled, not run on hardware
//
#include <hip/hip_runtime.h>

// ---------------------------------------------------------------------------
// BERT block for MI455X (gfx1250): bf16 WMMA GEMMs + flash attention.
// Compute-bound (~172 GFLOP vs ~150MB traffic) -> everything runs through
// v_wmma_f32_16x16x32_bf16 with f32 accumulation. GEMM tiles stream through
// GLOBAL_LOAD_ASYNC_TO_LDS_B128 (ASYNCcnt) with LDS double buffering so HBM/L2
// traffic overlaps the WMMA pipe.
// ---------------------------------------------------------------------------

typedef __bf16 bf16;
typedef __attribute__((ext_vector_type(16))) __bf16       bf16x16;
typedef __attribute__((ext_vector_type(8)))  float        f32x8;
typedef __attribute__((ext_vector_type(4)))  unsigned int u32x4;

union FragAB { u32x4 q[2]; bf16x16 v; };

#define WMMA_BF16(a, b, c) \
  __builtin_amdgcn_wmma_f32_16x16x32_bf16(false, (a), false, (b), (short)0, (c), false, false)

// ---- Async global->LDS copy (16B per lane, ASYNCcnt-tracked) --------------
// LDS operand is the low 32 bits of the generic address (ISA 10.2);
// inline asm keeps this portable across toolchain builtin arity.
static __device__ __forceinline__ void async_copy_b128(const bf16* gsrc, const bf16* ldst) {
  const unsigned lds_off = (unsigned)(size_t)ldst;
  asm volatile("global_load_async_to_lds_b128 %0, %1, off"
               :: "v"(lds_off), "v"(gsrc) : "memory");
}

// ---- Fragment loaders (LDS, K-contiguous rows) ----------------------------
// A 16x32 bf16 layout: lane holds row M=lane%16; element j -> K =
//   (j<8?0:16) + 8*(lane/16) + (j%8)  => two 16B loads, 32B apart.
static __device__ __forceinline__ bf16x16 load_frag_a(const bf16* base, int stride, int lane) {
  const int m  = lane & 15;
  const int kb = (lane >> 4) << 3;
  const bf16* p = base + m * stride + kb;
  FragAB f;
  f.q[0] = *reinterpret_cast<const u32x4*>(p);       // K = kb .. kb+7
  f.q[1] = *reinterpret_cast<const u32x4*>(p + 16);  // K = kb+16 .. kb+23
  return f.v;
}

// B 32x16 bf16 layout: lane holds col N=lane%16; element j -> K =
//   (j<8?0:16) + 8*(lane/16) + (j%8)  => two 16B loads, 32B apart.
static __device__ __forceinline__ bf16x16 load_frag_b(const bf16* base, int stride, int lane) {
  const int n  = lane & 15;
  const int kb = (lane >> 4) << 3;
  const bf16* p = base + n * stride + kb;
  FragAB f;
  f.q[0] = *reinterpret_cast<const u32x4*>(p);       // K = kb .. kb+7
  f.q[1] = *reinterpret_cast<const u32x4*>(p + 16);  // K = kb+16 .. kb+23
  return f.v;
}

// ---------------------------------------------------------------------------
// f32 -> bf16 convert (weights)
// ---------------------------------------------------------------------------
__global__ __launch_bounds__(256) void cvt_f32_bf16(const float* __restrict__ in,
                                                    bf16* __restrict__ out, int n) {
  int i = blockIdx.x * 256 + threadIdx.x;
  if (i < n) out[i] = (bf16)in[i];
}

// ---------------------------------------------------------------------------
// Mean-only layernorm: out_bf16 = w*(x - mean(x)) + b. One block per row.
// ---------------------------------------------------------------------------
__global__ __launch_bounds__(256) void ln_mean_kernel(const float* __restrict__ x,
                                                      const float* __restrict__ w,
                                                      const float* __restrict__ b,
                                                      bf16* __restrict__ out, int H) {
  __shared__ float red[256];
  const int row = blockIdx.x;
  const float* xr = x + (size_t)row * H;
  float s = 0.f;
  for (int i = threadIdx.x; i < H; i += 256) s += xr[i];
  red[threadIdx.x] = s;
  __syncthreads();
  for (int st = 128; st > 0; st >>= 1) {
    if (threadIdx.x < st) red[threadIdx.x] += red[threadIdx.x + st];
    __syncthreads();
  }
  const float mean = red[0] / (float)H;
  for (int i = threadIdx.x; i < H; i += 256)
    out[(size_t)row * H + i] = (bf16)(w[i] * (xr[i] - mean) + b[i]);
}

// ---------------------------------------------------------------------------
// GEMM: C[M,N] = A[M,K] @ W[N,K]^T (+bias)(+ReLU)(+residual)
// A,W bf16 K-contiguous. 256 threads = 8 waves; block tile 128x128, BK=64,
// double-buffered async global->LDS. Wave (wm 0..3, wn 0..1) owns 32x64.
// ---------------------------------------------------------------------------
template <bool RELU, bool RESID, bool OUT_BF16>
__global__ __launch_bounds__(256) void gemm_wmma(const bf16* __restrict__ A,
                                                 const bf16* __restrict__ W,
                                                 const float* __restrict__ bias,
                                                 const float* __restrict__ resid,
                                                 void* __restrict__ out,
                                                 int M, int N, int K) {
  __shared__ __align__(16) bf16 As[2][128][72];  // 128 rows x 64 K (+8 pad)
  __shared__ __align__(16) bf16 Bs[2][128][72];
  const int tid  = threadIdx.x;
  const int lane = tid & 31;
  const int wave = tid >> 5;
  const int wm = wave & 3;   // M strip (32 rows)
  const int wn = wave >> 2;  // N strip (64 cols)
  const int m0 = blockIdx.x * 128;
  const int n0 = blockIdx.y * 128;

  f32x8 acc[2][4] = {};

  // 8 async b128 instructions per tile per wave (4 A + 4 B).
  auto issue_tile = [&](int k0, int b) {
#pragma unroll
    for (int i = 0; i < 4; ++i) {
      const int idx = tid + i * 256;   // 0..1023
      const int row = idx >> 3;        // 0..127
      const int col = (idx & 7) << 3;  // 0..56
      async_copy_b128(&A[(size_t)(m0 + row) * K + k0 + col], &As[b][row][col]);
      async_copy_b128(&W[(size_t)(n0 + row) * K + k0 + col], &Bs[b][row][col]);
    }
  };

  auto compute = [&](int b) {
    bf16x16 afr[2][2];
#pragma unroll
    for (int mt = 0; mt < 2; ++mt)
#pragma unroll
      for (int kk = 0; kk < 2; ++kk)
        afr[mt][kk] = load_frag_a(&As[b][wm * 32 + mt * 16][kk * 32], 72, lane);
#pragma unroll
    for (int nt = 0; nt < 4; ++nt)
#pragma unroll
      for (int kk = 0; kk < 2; ++kk) {
        bf16x16 bfr = load_frag_b(&Bs[b][wn * 64 + nt * 16][kk * 32], 72, lane);
        acc[0][nt] = WMMA_BF16(afr[0][kk], bfr, acc[0][nt]);
        acc[1][nt] = WMMA_BF16(afr[1][kk], bfr, acc[1][nt]);
      }
  };

  const int nk = K >> 6;  // K / 64; all K here are multiples of 64
  issue_tile(0, 0);
  for (int k = 0; k < nk - 1; ++k) {
    issue_tile((k + 1) << 6, (k + 1) & 1);
    // tile k's 8 async loads complete when <= 8 (next tile's) remain; loads
    // retire in order.
    asm volatile("s_wait_asynccnt 0x8" ::: "memory");
    __syncthreads();
    compute(k & 1);
    __syncthreads();  // everyone done reading buf before it is refilled
  }
  asm volatile("s_wait_asynccnt 0x0" ::: "memory");
  __syncthreads();
  compute((nk - 1) & 1);

  // Epilogue. C layout: element r -> row M = r + 8*(lane/16), col N = lane%16.
  const int rb = (lane >> 4) << 3;
#pragma unroll
  for (int mt = 0; mt < 2; ++mt) {
#pragma unroll
    for (int nt = 0; nt < 4; ++nt) {
      const int col = n0 + wn * 64 + nt * 16 + (lane & 15);
      const float bv = bias[col];
#pragma unroll
      for (int r = 0; r < 8; ++r) {
        const int row = m0 + wm * 32 + mt * 16 + rb + r;
        float v = acc[mt][nt][r] + bv;
        if (RELU) v = v > 0.f ? v : 0.f;
        if (RESID) v += resid[(size_t)row * N + col];
        if (OUT_BF16)
          ((bf16*)out)[(size_t)row * N + col] = (bf16)v;
        else
          ((float*)out)[(size_t)row * N + col] = v;
      }
    }
  }
}

// ---------------------------------------------------------------------------
// Flash attention. Grid: (S/64, HEADS). 128 threads = 4 waves.
// Each wave owns a 16-row query strip; loop over 64-wide key blocks with
// online softmax. scores = Q K^T / 8 via WMMA; ctx += P V via WMMA.
// K tile streams in via async-to-LDS; V is transposed through VGPRs.
// ---------------------------------------------------------------------------
__global__ __launch_bounds__(128) void attn_flash_kernel(const bf16* __restrict__ Qg,
                                                         const bf16* __restrict__ Kg,
                                                         const bf16* __restrict__ Vg,
                                                         bf16* __restrict__ ctx,
                                                         int S, int INNER_) {
  __shared__ __align__(16) bf16 Qs[64][72];       // [q][d]
  __shared__ __align__(16) bf16 Ks[64][72];       // [t][d]  (== B layout of K^T)
  __shared__ __align__(16) bf16 Vt[64][72];       // [d][t]  (transposed V)
  __shared__ __align__(16) bf16 Ps[4][16][72];    // per-wave P re-fragment patch

  const int tid  = threadIdx.x;
  const int lane = tid & 31;
  const int wave = tid >> 5;
  const int qbase = blockIdx.x * 64;
  const int hoff  = blockIdx.y * 64;

  // Load 64x64 Q tile once.
#pragma unroll
  for (int i = 0; i < 4; ++i) {
    const int idx = tid + i * 128;       // 0..511
    const int row = idx >> 3;            // 0..63
    const int col = (idx & 7) << 3;      // 0..56
    *reinterpret_cast<u32x4*>(&Qs[row][col]) =
        *reinterpret_cast<const u32x4*>(&Qg[(size_t)(qbase + row) * INNER_ + hoff + col]);
  }
  __syncthreads();

  bf16x16 aQ0 = load_frag_a(&Qs[wave * 16][0], 72, lane);   // K(d)=0..31
  bf16x16 aQ1 = load_frag_a(&Qs[wave * 16][32], 72, lane);  // K(d)=32..63

  f32x8 cacc[4] = {};
  float mrow[8], lrow[8];
#pragma unroll
  for (int r = 0; r < 8; ++r) { mrow[r] = -1e30f; lrow[r] = 0.f; }

  for (int t0 = 0; t0 < S; t0 += 64) {
    __syncthreads();  // protect Ks/Vt reuse across iterations
    // K tile: async global->LDS (4 b128 per lane).
#pragma unroll
    for (int i = 0; i < 4; ++i) {
      const int idx = tid + i * 128;
      const int row = idx >> 3;
      const int col = (idx & 7) << 3;
      async_copy_b128(&Kg[(size_t)(t0 + row) * INNER_ + hoff + col], &Ks[row][col]);
    }
    // V tile: load + transpose through VGPRs.
#pragma unroll
    for (int i = 0; i < 4; ++i) {
      const int idx = tid + i * 128;
      const int row = idx >> 3;
      const int col = (idx & 7) << 3;
      u32x4 vv = *reinterpret_cast<const u32x4*>(&Vg[(size_t)(t0 + row) * INNER_ + hoff + col]);
      const bf16* vb = reinterpret_cast<const bf16*>(&vv);
#pragma unroll
      for (int j = 0; j < 8; ++j) Vt[col + j][row] = vb[j];  // transpose store
    }
    asm volatile("s_wait_asynccnt 0x0" ::: "memory");
    __syncthreads();

    // ---- scores: 4 tiles of 16x16, K(d)=64 -> 2 WMMAs each ----
    f32x8 sacc[4] = {};
#pragma unroll
    for (int nt = 0; nt < 4; ++nt) {
      bf16x16 b0 = load_frag_b(&Ks[nt * 16][0], 72, lane);
      bf16x16 b1 = load_frag_b(&Ks[nt * 16][32], 72, lane);
      sacc[nt] = WMMA_BF16(aQ0, b0, sacc[nt]);
      sacc[nt] = WMMA_BF16(aQ1, b1, sacc[nt]);
    }
#pragma unroll
    for (int nt = 0; nt < 4; ++nt)
#pragma unroll
      for (int r = 0; r < 8; ++r) sacc[nt][r] *= 0.125f;  // 1/sqrt(64)

    // ---- online softmax. Row M = r + 8*(lane/16) lives across a 16-lane
    // group in element r, so row reductions are shfl_xor over masks 1..8. ----
    float alpha[8];
#pragma unroll
    for (int r = 0; r < 8; ++r) {
      float mx = fmaxf(fmaxf(sacc[0][r], sacc[1][r]), fmaxf(sacc[2][r], sacc[3][r]));
#pragma unroll
      for (int off = 1; off < 16; off <<= 1) mx = fmaxf(mx, __shfl_xor(mx, off, 32));
      const float mn = fmaxf(mrow[r], mx);
      alpha[r] = __expf(mrow[r] - mn);
      mrow[r]  = mn;
    }
    float rs[8];
#pragma unroll
    for (int r = 0; r < 8; ++r) rs[r] = 0.f;
#pragma unroll
    for (int nt = 0; nt < 4; ++nt)
#pragma unroll
      for (int r = 0; r < 8; ++r) {
        const float p = __expf(sacc[nt][r] - mrow[r]);
        sacc[nt][r] = p;
        rs[r] += p;
      }
#pragma unroll
    for (int r = 0; r < 8; ++r) {
      float t = rs[r];
#pragma unroll
      for (int off = 1; off < 16; off <<= 1) t += __shfl_xor(t, off, 32);
      lrow[r] = lrow[r] * alpha[r] + t;
    }
#pragma unroll
    for (int dt = 0; dt < 4; ++dt)
#pragma unroll
      for (int r = 0; r < 8; ++r) cacc[dt][r] *= alpha[r];

    // ---- C-layout -> A-layout via wave-private LDS patch ----
    const int rb = (lane >> 4) << 3;
#pragma unroll
    for (int nt = 0; nt < 4; ++nt)
#pragma unroll
      for (int r = 0; r < 8; ++r)
        Ps[wave][rb + r][nt * 16 + (lane & 15)] = (bf16)sacc[nt][r];
    asm volatile("s_wait_dscnt 0" ::: "memory");  // intra-wave LDS RAW

    bf16x16 aP0 = load_frag_a(&Ps[wave][0][0], 72, lane);   // K(t)=0..31
    bf16x16 aP1 = load_frag_a(&Ps[wave][0][32], 72, lane);  // K(t)=32..63

    // ---- ctx += P @ V ----
#pragma unroll
    for (int dt = 0; dt < 4; ++dt) {
      bf16x16 b0 = load_frag_b(&Vt[dt * 16][0], 72, lane);
      bf16x16 b1 = load_frag_b(&Vt[dt * 16][32], 72, lane);
      cacc[dt] = WMMA_BF16(aP0, b0, cacc[dt]);
      cacc[dt] = WMMA_BF16(aP1, b1, cacc[dt]);
    }
  }

  // finalize: ctx /= l, write bf16
  const int rb = (lane >> 4) << 3;
#pragma unroll
  for (int dt = 0; dt < 4; ++dt)
#pragma unroll
    for (int r = 0; r < 8; ++r) {
      const int row = qbase + wave * 16 + rb + r;
      const int col = hoff + dt * 16 + (lane & 15);
      ctx[(size_t)row * INNER_ + col] = (bf16)(cacc[dt][r] / lrow[r]);
    }
}

// ---------------------------------------------------------------------------
// Host orchestration
// ---------------------------------------------------------------------------
extern "C" void kernel_launch(void* const* d_in, const int* in_sizes, int n_in,
                              void* d_out, int out_size, void* d_ws, size_t ws_size,
                              hipStream_t stream) {
  constexpr int S = 4096, H = 1024, I = 4096, NN = 1024;  // NN = HEADS*HEAD_DIM

  const float* x     = (const float*)d_in[0];
  const float* an_w  = (const float*)d_in[1];
  const float* an_b  = (const float*)d_in[2];
  const float* q_w   = (const float*)d_in[3];
  const float* q_b   = (const float*)d_in[4];
  const float* k_w   = (const float*)d_in[5];
  const float* k_b   = (const float*)d_in[6];
  const float* v_w   = (const float*)d_in[7];
  const float* v_b   = (const float*)d_in[8];
  const float* o_w   = (const float*)d_in[9];
  const float* o_b   = (const float*)d_in[10];
  const float* fn_w  = (const float*)d_in[11];
  const float* fn_b  = (const float*)d_in[12];
  const float* ff1_w = (const float*)d_in[13];
  const float* ff1_b = (const float*)d_in[14];
  const float* ff2_w = (const float*)d_in[15];
  const float* ff2_b = (const float*)d_in[16];

  // Workspace carve-up (~120 MB total).
  char* ws = (char*)d_ws;
  size_t off = 0;
  auto carve = [&](size_t bytes) {
    char* p = ws + off;
    off += (bytes + 255) & ~(size_t)255;
    return p;
  };
  bf16* Wq  = (bf16*)carve((size_t)NN * H * 2);
  bf16* Wk  = (bf16*)carve((size_t)NN * H * 2);
  bf16* Wv  = (bf16*)carve((size_t)NN * H * 2);
  bf16* Wo  = (bf16*)carve((size_t)H * NN * 2);
  bf16* W1  = (bf16*)carve((size_t)I * H * 2);
  bf16* W2  = (bf16*)carve((size_t)H * I * 2);
  bf16* xn  = (bf16*)carve((size_t)S * H * 2);
  bf16* Qb  = (bf16*)carve((size_t)S * NN * 2);
  bf16* Kb  = (bf16*)carve((size_t)S * NN * 2);
  bf16* Vb  = (bf16*)carve((size_t)S * NN * 2);
  bf16* Cb  = (bf16*)carve((size_t)S * NN * 2);
  float* x2 = (float*)carve((size_t)S * H * 4);
  bf16* xn2 = (bf16*)carve((size_t)S * H * 2);
  bf16* hb  = (bf16*)carve((size_t)S * I * 2);
  (void)ws_size; (void)in_sizes; (void)n_in; (void)out_size;

  // 1) weights -> bf16
  cvt_f32_bf16<<<(NN * H + 255) / 256, 256, 0, stream>>>(q_w, Wq, NN * H);
  cvt_f32_bf16<<<(NN * H + 255) / 256, 256, 0, stream>>>(k_w, Wk, NN * H);
  cvt_f32_bf16<<<(NN * H + 255) / 256, 256, 0, stream>>>(v_w, Wv, NN * H);
  cvt_f32_bf16<<<(H * NN + 255) / 256, 256, 0, stream>>>(o_w, Wo, H * NN);
  cvt_f32_bf16<<<(I * H + 255) / 256, 256, 0, stream>>>(ff1_w, W1, I * H);
  cvt_f32_bf16<<<(H * I + 255) / 256, 256, 0, stream>>>(ff2_w, W2, H * I);

  // 2) LN1
  ln_mean_kernel<<<S, 256, 0, stream>>>(x, an_w, an_b, xn, H);

  // 3) Q/K/V projections
  dim3 gP(S / 128, NN / 128);
  gemm_wmma<false, false, true><<<gP, 256, 0, stream>>>(xn, Wq, q_b, nullptr, Qb, S, NN, H);
  gemm_wmma<false, false, true><<<gP, 256, 0, stream>>>(xn, Wk, k_b, nullptr, Kb, S, NN, H);
  gemm_wmma<false, false, true><<<gP, 256, 0, stream>>>(xn, Wv, v_b, nullptr, Vb, S, NN, H);

  // 4) flash attention
  dim3 gA(S / 64, 16);
  attn_flash_kernel<<<gA, 128, 0, stream>>>(Qb, Kb, Vb, Cb, S, NN);

  // 5) O projection + residual (f32)
  dim3 gO(S / 128, H / 128);
  gemm_wmma<false, true, false><<<gO, 256, 0, stream>>>(Cb, Wo, o_b, x, x2, S, H, NN);

  // 6) LN2
  ln_mean_kernel<<<S, 256, 0, stream>>>(x2, fn_w, fn_b, xn2, H);

  // 7) FF1 + ReLU (bf16)
  dim3 gF1(S / 128, I / 128);
  gemm_wmma<true, false, true><<<gF1, 256, 0, stream>>>(xn2, W1, ff1_b, nullptr, hb, S, I, H);

  // 8) FF2 + residual -> f32 output
  dim3 gF2(S / 128, H / 128);
  gemm_wmma<false, true, false><<<gF2, 256, 0, stream>>>(hb, W2, ff2_b, x2, (float*)d_out, S, H, I);
}
